// DeltaNet_31877247271575
// MI455X (gfx1250) — matrix operand order repaired and checked by reading, not compiled, not run on hardware
//
#include <hip/hip_runtime.h>
#include <hip/hip_bf16.h>

// ---------------- problem constants ----------------
#define B_   4
#define L_   4096
#define D_   1024
#define H_   4
#define DKH  256         // head dim (dk == dv == 256)
#define CH   32          // chunk length
#define NCH  (L_/CH)     // 128 chunks
#define BLT  (B_*L_)     // 16384 tokens

// ---------------- CDNA5 feature detection (compile-robust) ----------------
typedef __attribute__((ext_vector_type(4))) unsigned int u32x4;
typedef __attribute__((ext_vector_type(8))) int i32x8;
typedef __attribute__((ext_vector_type(4))) int i32x4;

#if __has_builtin(__builtin_amdgcn_tensor_load_to_lds) && __has_builtin(__builtin_amdgcn_s_wait_tensorcnt)
#define HAVE_TDM 1
#endif
#if __has_builtin(__builtin_amdgcn_global_load_async_to_lds_b128) && __has_builtin(__builtin_amdgcn_s_wait_asynccnt)
#define HAVE_ASYNC 1
#endif

// Low 32 bits of a generic shared-aperture address are the LDS byte offset
__device__ __forceinline__ unsigned lds_off(const void* p) {
  return (unsigned)(unsigned long long)p;
}

#ifdef HAVE_TDM
// Build a 2D-tile Tensor DMA descriptor (D#) and issue TENSOR_LOAD_TO_LDS.
// Fields packed per CDNA5 ISA 8.3 (group0) / 8.4 (group1).
//   dsz_code: 0=1B,1=2B,2=4B ; dims/strides in data_size units.
//   pad_*: LDS padding (pad_interval code: 5 => every 64 DWORDs, pad_amount code: 0 => 1 DWORD)
__device__ __forceinline__ void tdm_load_2d(unsigned lds_addr, const void* gsrc,
                                            unsigned dsz_code, unsigned tensor_d0,
                                            unsigned tile_d0, unsigned rows,
                                            unsigned row_stride, unsigned pad_en,
                                            unsigned pad_interval, unsigned pad_amount) {
  unsigned long long ga = (unsigned long long)gsrc;
  u32x4 g0;
  g0[0] = 1u;                                              // count=1, user descriptor
  g0[1] = lds_addr;                                        // lds_addr[31:0]
  g0[2] = (unsigned)(ga & 0xFFFFFFFFu);                    // global_addr[31:0]
  g0[3] = (unsigned)((ga >> 32) & 0x1FFFFFFu) | (2u << 30);// global_addr[56:32] | type=2
  i32x8 g1;
  g1[0] = (int)((dsz_code << 16) | (pad_en ? (1u << 20) : 0u) |
                ((pad_interval & 7u) << 22) | ((pad_amount & 127u) << 25));
  g1[1] = (int)((tensor_d0 & 0xFFFFu) << 16);              // tensor_dim0[15:0] -> bits 63:48
  g1[2] = (int)(((tensor_d0 >> 16) & 0xFFFFu) |            // tensor_dim0[31:16]
                ((rows & 0xFFFFu) << 16));                 // tensor_dim1[15:0]
  g1[3] = (int)(((rows >> 16) & 0xFFFFu) |                 // tensor_dim1[31:16]
                ((tile_d0 & 0xFFFFu) << 16));              // tile_dim0
  g1[4] = (int)(rows & 0xFFFFu);                           // tile_dim1 (tile_dim2 = 0)
  g1[5] = (int)row_stride;                                 // tensor_dim0_stride[31:0]
  g1[6] = 0;
  g1[7] = 0;
  i32x4 z4 = {0, 0, 0, 0};
#if __clang_major__ >= 23
  i32x8 z8 = {0, 0, 0, 0, 0, 0, 0, 0};
  __builtin_amdgcn_tensor_load_to_lds(g0, g1, z4, z4, z8, 0);
#else
  __builtin_amdgcn_tensor_load_to_lds(g0, g1, z4, z4, 0);
#endif
}
#endif

// ---------------- WMMA types / helpers (gfx1250, wave32) ----------------
typedef __attribute__((ext_vector_type(16))) __bf16 bf16x16;
typedef __attribute__((ext_vector_type(8)))  float  v8f;

__device__ __forceinline__ int lane_id() { return threadIdx.x & 31; }
__device__ __forceinline__ __bf16 f2bf(float f) { return (__bf16)f; }

__device__ __forceinline__ v8f zero8() {
  v8f z;
#pragma unroll
  for (int i = 0; i < 8; ++i) z[i] = 0.f;
  return z;
}

__device__ __forceinline__ v8f wmma_bf16(bf16x16 a, bf16x16 b, v8f c) {
  // D = A(16x32) * B(32x16) + C, fp32 accumulate
  return __builtin_amdgcn_wmma_f32_16x16x32_bf16(false, a, false, b, (short)0, c,
                                                 false, false);
}

// 16 contiguous bf16 (32B, 16B-aligned) -> fragment register pair
__device__ __forceinline__ bf16x16 ld_frag16(const __bf16* p) {
  union { uint4 u[2]; bf16x16 v; } cv;
  cv.u[0] = ((const uint4*)p)[0];
  cv.u[1] = ((const uint4*)p)[1];
  return cv.v;
}

// A fragment (16x32, MxK) from row-major bf16 [m][k].
// Lanes 0-15 -> M=lane, K in {0..15}; lanes 16-31 -> M=lane-16, K in {16..31}.
__device__ __forceinline__ bf16x16 ld_a(const __bf16* p, int stride) {
  int l = lane_id(); int m = l & 15; int kb = (l < 16) ? 0 : 16;
  bf16x16 r;
#pragma unroll
  for (int e = 0; e < 16; ++e) { int k = e + kb; r[e] = p[m * stride + k]; }
  return r;
}
// A = X^T fragment: A[m][k] = X[k*stride + c0 + m]   (k = 0..31)
__device__ __forceinline__ bf16x16 ld_at(const __bf16* X, int c0, int stride) {
  int l = lane_id(); int m = l & 15; int kb = (l < 16) ? 0 : 16;
  bf16x16 r;
#pragma unroll
  for (int e = 0; e < 16; ++e) { int k = e + kb; r[e] = X[k * stride + c0 + m]; }
  return r;
}
// B fragment (32x16, KxN) from COLUMN-major staging: B[k][n] = p[n*stride + k].
// Per-lane data is contiguous -> two ds_load_b128.
__device__ __forceinline__ bf16x16 ld_bcm(const __bf16* p, int stride) {
  int l = lane_id(); int n = l & 15; int kb = (l < 16) ? 0 : 16;
  return ld_frag16(p + n * stride + kb);
}
// B fragment (32x16) from row-major [k][n] (strided gather fallback).
__device__ __forceinline__ bf16x16 ld_b(const __bf16* p, int stride) {
  int l = lane_id(); int n = l & 15; int kb = (l < 16) ? 0 : 16;
  bf16x16 r;
#pragma unroll
  for (int e = 0; e < 16; ++e) r[e] = p[(kb + e) * stride + n];
  return r;
}
// B = X^T fragment: B[k][n] = X[(n0+n)*stride + k0 + k] (contiguous per lane)
__device__ __forceinline__ bf16x16 ld_bt(const __bf16* X, int n0, int k0, int stride) {
  int l = lane_id(); int n = l & 15; int kb = (l < 16) ? 0 : 16;
  return ld_frag16(X + (n0 + n) * stride + k0 + kb);
}
// B fragment from f32 row-major (convert + optional negate); for LDS-resident fp32 state S.
__device__ __forceinline__ bf16x16 ld_b_f32(const float* p, int stride, float sgn) {
  int l = lane_id(); int n = l & 15; int kb = (l < 16) ? 0 : 16;
  bf16x16 r;
#pragma unroll
  for (int e = 0; e < 16; ++e) r[e] = f2bf(sgn * p[(kb + e) * stride + n]);
  return r;
}
// C/D fragment (16x16 f32): VGPR r -> row (r + 0|8), lane -> column.
__device__ __forceinline__ v8f ld_c(const float* p, int stride) {
  int l = lane_id(); int n = l & 15; int mb = (l < 16) ? 0 : 8;
  v8f r;
#pragma unroll
  for (int e = 0; e < 8; ++e) r[e] = p[(mb + e) * stride + n];
  return r;
}
__device__ __forceinline__ void st_c(float* p, int stride, v8f d) {
  int l = lane_id(); int n = l & 15; int mb = (l < 16) ? 0 : 8;
#pragma unroll
  for (int e = 0; e < 8; ++e) p[(size_t)(mb + e) * stride + n] = d[e];
}
__device__ __forceinline__ void st_c_bf16(__bf16* p, int stride, v8f d) {
  int l = lane_id(); int n = l & 15; int mb = (l < 16) ? 0 : 8;
#pragma unroll
  for (int e = 0; e < 8; ++e) p[(size_t)(mb + e) * stride + n] = f2bf(d[e]);
}
// Store C/D fragment transposed (column-major): element (m,n) -> p[n*stride + m]
__device__ __forceinline__ void st_cm_bf16(__bf16* p, int stride, v8f d) {
  int l = lane_id(); int n = l & 15; int mb = (l < 16) ? 0 : 8;
#pragma unroll
  for (int e = 0; e < 8; ++e) p[n * stride + mb + e] = f2bf(d[e]);
}

__device__ __forceinline__ float block_sum(float v, float* buf) {
  int t = threadIdx.x;
  buf[t] = v;
  __syncthreads();
  for (int s = 128; s > 0; s >>= 1) {
    if (t < s) buf[t] += buf[t + s];
    __syncthreads();
  }
  float r = buf[0];
  __syncthreads();
  return r;
}

// ---------------- kernel: f32 -> bf16 convert ----------------
__global__ __launch_bounds__(256) void cvt_bf16_k(const float* __restrict__ x,
                                                  __bf16* __restrict__ y, int n) {
  int i = (blockIdx.x * 256 + threadIdx.x) * 4;
  if (i < n) {
    float4 v = *(const float4*)(x + i);
    y[i + 0] = f2bf(v.x); y[i + 1] = f2bf(v.y);
    y[i + 2] = f2bf(v.z); y[i + 3] = f2bf(v.w);
  }
}

// ---------------- kernel: WMMA GEMM  C[MxN] = A[MxK](bf16) @ W[NxK]^T ----------------
#define BM 128
#define BN 128
#define BK 32
#define BKp 40   // padded row stride of As
#define BNK 40   // padded column stride of Bs (column-major: Bs[n*BNK + k])
__global__ __launch_bounds__(256) void gemm_bf16f32(const __bf16* __restrict__ A,
                                                    const float* __restrict__ W,
                                                    float* __restrict__ C,
                                                    int M, int N, int K) {
  __shared__ __align__(16) __bf16 As[BM * BKp];
  __shared__ __align__(16) __bf16 Bs[BN * BNK];
  int m0 = blockIdx.y * BM, n0 = blockIdx.x * BN;
  int tid = threadIdx.x;
  int wave = tid >> 5;
  int wr = wave >> 1, wc = wave & 1;   // 4x2 wave grid; each wave: 32x64 output
  v8f acc[2][4];
#pragma unroll
  for (int i = 0; i < 2; ++i)
#pragma unroll
    for (int j = 0; j < 4; ++j) acc[i][j] = zero8();

  int nK = K / BK;
  for (int kt = 0; kt < nK; ++kt) {
    int k0 = kt * BK;
    __syncthreads();
    {  // stage A tile (128x32 bf16), 32B per thread
      int r = tid >> 1, off = (tid & 1) * 16;
      const uint4* src = (const uint4*)(A + (size_t)(m0 + r) * K + k0 + off);
      uint4* dst = (uint4*)(As + r * BKp + off);
      dst[0] = src[0]; dst[1] = src[1];
    }
    {  // stage B tile COLUMN-major: Bs[n*BNK + k] = W[(n0+n)*K + k0+k]
      int n = tid >> 1, koff = (tid & 1) * 16;
      const float* src = W + (size_t)(n0 + n) * K + k0 + koff;
      __bf16* dst = Bs + n * BNK + koff;
#pragma unroll
      for (int i = 0; i < 16; ++i) dst[i] = f2bf(src[i]);
    }
    __syncthreads();
    if (kt + 1 < nK) {  // global_prefetch_b8 of next A tile
      __builtin_prefetch(A + (size_t)(m0 + (tid >> 1)) * K + k0 + BK, 0, 1);
    }
    bf16x16 af[2], bfrag[4];
#pragma unroll
    for (int tm = 0; tm < 2; ++tm) af[tm] = ld_a(As + (wr * 32 + tm * 16) * BKp, BKp);
#pragma unroll
    for (int tn = 0; tn < 4; ++tn) bfrag[tn] = ld_bcm(Bs + (wc * 64 + tn * 16) * BNK, BNK);
#pragma unroll
    for (int tm = 0; tm < 2; ++tm)
#pragma unroll
      for (int tn = 0; tn < 4; ++tn) acc[tm][tn] = wmma_bf16(af[tm], bfrag[tn], acc[tm][tn]);
  }
#pragma unroll
  for (int tm = 0; tm < 2; ++tm)
#pragma unroll
    for (int tn = 0; tn < 4; ++tn) {
      float* cp = C + (size_t)(m0 + wr * 32 + tm * 16) * N + n0 + wc * 64 + tn * 16;
      st_c(cp, N, acc[tm][tn]);
    }
}

// ---------------- kernel: causal depthwise conv (K=4) + SiLU ----------------
#define CSEG 128
__global__ __launch_bounds__(256) void conv_silu_k(const float* __restrict__ xin,
                                                   const float* __restrict__ wc,
                                                   float* __restrict__ out) {
  int ch = blockIdx.x * 256 + threadIdx.x;
  int b = blockIdx.z;
  int l0 = blockIdx.y * CSEG;
  float w0 = wc[ch * 4 + 0], w1 = wc[ch * 4 + 1], w2 = wc[ch * 4 + 2], w3 = wc[ch * 4 + 3];
  size_t base = (size_t)b * L_ * D_ + ch;
  float xm3 = (l0 >= 3) ? xin[base + (size_t)(l0 - 3) * D_] : 0.f;
  float xm2 = (l0 >= 2) ? xin[base + (size_t)(l0 - 2) * D_] : 0.f;
  float xm1 = (l0 >= 1) ? xin[base + (size_t)(l0 - 1) * D_] : 0.f;
  for (int l = l0; l < l0 + CSEG; ++l) {
    float xc = xin[base + (size_t)l * D_];
    float y = w0 * xm3 + w1 * xm2 + w2 * xm1 + w3 * xc;
    out[base + (size_t)l * D_] = y / (1.f + __expf(-y));   // silu
    xm3 = xm2; xm2 = xm1; xm1 = xc;
  }
}

// ---------------- kernel: beta & mix gates ----------------
__global__ __launch_bounds__(128) void betamix_k(const float* __restrict__ x,
                                                 const float* __restrict__ Wb,
                                                 const float* __restrict__ Wmix,
                                                 float* __restrict__ beta,
                                                 float* __restrict__ mixo) {
  __shared__ float red[128 * 8];
  int token = blockIdx.x, t = threadIdx.x;
  float acc[8];
#pragma unroll
  for (int j = 0; j < 8; ++j) acc[j] = 0.f;
  const float* xr = x + (size_t)token * D_;
  for (int k = t; k < D_; k += 128) {
    float xv = xr[k];
#pragma unroll
    for (int hh = 0; hh < 4; ++hh) {
      acc[hh]     += xv * Wb[hh * D_ + k];
      acc[4 + hh] += xv * Wmix[hh * D_ + k];
    }
  }
#pragma unroll
  for (int j = 0; j < 8; ++j) red[t * 8 + j] = acc[j];
  __syncthreads();
  for (int s = 64; s > 0; s >>= 1) {
    if (t < s)
#pragma unroll
      for (int j = 0; j < 8; ++j) red[t * 8 + j] += red[(t + s) * 8 + j];
    __syncthreads();
  }
  if (t < 4) beta[token * 4 + t] = 1.f / (1.f + __expf(-red[t]));
  else if (t < 8) mixo[token * 4 + (t - 4)] = 1.f / (1.f + __expf(-red[t]));
}

// ---------------- kernel: l2norm + beta-scale, split to (b,h,l,d) bf16 ----------------
__global__ __launch_bounds__(256) void norm_split_k(const float* __restrict__ qc,
                                                    const float* __restrict__ kc,
                                                    const float* __restrict__ vc,
                                                    const float* __restrict__ beta,
                                                    __bf16* __restrict__ qn,
                                                    __bf16* __restrict__ knn,
                                                    __bf16* __restrict__ kbb,
                                                    __bf16* __restrict__ vbb) {
  __shared__ float buf[256];
  int token = blockIdx.x;
  int b = token / L_, l = token % L_;
  int d = threadIdx.x;
#pragma unroll
  for (int hh = 0; hh < H_; ++hh) {
    size_t gi = (size_t)token * D_ + hh * DKH + d;
    float qv = qc[gi], kv = kc[gi], vv = vc[gi];
    float sq = block_sum(qv * qv, buf);
    float sk = block_sum(kv * kv, buf);
    float rq = rsqrtf(sq + 1e-6f), rk = rsqrtf(sk + 1e-6f);
    float bet = beta[token * 4 + hh];
    size_t oi = (((size_t)(b * H_ + hh)) * L_ + l) * DKH + d;
    qn[oi]  = f2bf(qv * rq);
    knn[oi] = f2bf(kv * rk);
    kbb[oi] = f2bf(kv * rk * bet);
    vbb[oi] = f2bf(vv * bet);
  }
}

// ---------------- kernel: intra-chunk UT transform, u = T@v, w = T@k_beta ----------------
__global__ __launch_bounds__(256) void intra_chunk_k(const __bf16* __restrict__ kn,
                                                     const __bf16* __restrict__ kb,
                                                     const __bf16* __restrict__ vb,
                                                     float* __restrict__ u,
                                                     __bf16* __restrict__ w) {
  __shared__ __align__(16) __bf16 sk[CH * DKH];
  __shared__ __align__(16) __bf16 skb[CH * DKH];
  __shared__ __align__(16) __bf16 sv[CH * DKH];
  __shared__ float sattn[CH * 33];
  __shared__ __bf16 sT[CH * CH];
  int bh = blockIdx.y, ck = blockIdx.x;
  size_t gb = ((size_t)bh * L_ + (size_t)ck * CH) * DKH;
  int tid = threadIdx.x, wave = tid >> 5, lane = tid & 31;
#ifdef HAVE_ASYNC
  {  // ASYNCcnt path: global -> LDS without VGPR round-trip
#pragma unroll
    for (int i = 0; i < 4; ++i) {
      int idx = tid + i * 256;
      __builtin_amdgcn_global_load_async_to_lds_b128((i32x4*)((const uint4*)(kn + gb) + idx),
                                                     (i32x4*)((uint4*)sk + idx), 0, 0);
      __builtin_amdgcn_global_load_async_to_lds_b128((i32x4*)((const uint4*)(kb + gb) + idx),
                                                     (i32x4*)((uint4*)skb + idx), 0, 0);
      __builtin_amdgcn_global_load_async_to_lds_b128((i32x4*)((const uint4*)(vb + gb) + idx),
                                                     (i32x4*)((uint4*)sv + idx), 0, 0);
    }
    __builtin_amdgcn_s_wait_asynccnt(0);
  }
#else
  {  // stage the three 32x256 bf16 chunk tiles
    const uint4* gk  = (const uint4*)(kn + gb);
    const uint4* gkb = (const uint4*)(kb + gb);
    const uint4* gv  = (const uint4*)(vb + gb);
    uint4* lk = (uint4*)sk; uint4* lkb = (uint4*)skb; uint4* lv = (uint4*)sv;
#pragma unroll
    for (int i = 0; i < 4; ++i) {
      int idx = tid + i * 256;
      lk[idx] = gk[idx]; lkb[idx] = gkb[idx]; lv[idx] = gv[idx];
    }
  }
#endif
  __syncthreads();
  // attn = -(k_beta @ k^T) on strict lower triangle (4 tiles, waves 0-3, K=256)
  if (wave < 4) {
    int tm = wave >> 1, tn = wave & 1;
    v8f acc = zero8();
#pragma unroll
    for (int ks = 0; ks < 8; ++ks) {
      bf16x16 a = ld_a(skb + (tm * 16) * DKH + ks * 32, DKH);
      bf16x16 b = ld_bt(sk, tn * 16, ks * 32, DKH);
      acc = wmma_bf16(a, b, acc);
    }
    int n = tn * 16 + (lane & 15), mb = (lane < 16) ? 0 : 8;
#pragma unroll
    for (int e = 0; e < 8; ++e) {
      int m = tm * 16 + mb + e;
      sattn[m * 33 + n] = (m > n) ? -acc[e] : 0.f;
    }
  }
  __syncthreads();
  // forward substitution: row i += row_i @ attn  (wave-0 lanes in lockstep)
  if (tid < 32) {
    int col = tid;
    for (int i = 1; i < CH; ++i) {
      float s = 0.f;
      for (int j = 0; j < i; ++j) s += sattn[i * 33 + j] * sattn[j * 33 + col];
      sattn[i * 33 + col] += s;
    }
  }
  __syncthreads();
  for (int idx = tid; idx < CH * CH; idx += 256) {
    int m = idx >> 5, n = idx & 31;
    sT[m * CH + n] = f2bf(sattn[m * 33 + n] + ((m == n) ? 1.f : 0.f));
  }
  __syncthreads();
  // u = T @ v_beta (tiles 0..31), w = T @ k_beta (tiles 32..63); 8 tiles per wave
#pragma unroll
  for (int t = 0; t < 8; ++t) {
    int tile = wave * 8 + t;
    int isw = tile >> 5, id2 = tile & 31;
    int tm = id2 >> 4, tn = id2 & 15;
    bf16x16 a = ld_a(sT + (tm * 16) * CH, CH);
    bf16x16 b;
    if (isw) b = ld_b(skb + tn * 16, DKH);
    else     b = ld_b(sv + tn * 16, DKH);
    v8f d = wmma_bf16(a, b, zero8());
    size_t ob = gb + (size_t)(tm * 16) * DKH + tn * 16;
    if (isw) st_c_bf16(w + ob, DKH, d);
    else     st_c(u + ob, DKH, d);
  }
}

// ---------------- kernel: sequential inter-chunk scan (S in LDS, fp32) ----------------
#define SS 65      // padded stride for the 256 x 64 fp32 state slice (TDM pad produces this)
#define SUNK 40    // padded column stride of bf16 u_new staging (column-major)
__global__ __launch_bounds__(256) void scan_chunks_k(const __bf16* __restrict__ qn,
                                                     const __bf16* __restrict__ kn,
                                                     const __bf16* __restrict__ w,
                                                     const float* __restrict__ u,
                                                     float* __restrict__ o) {
  extern __shared__ __align__(16) char smem[];
  float*  S   = (float*)smem;                    // 256 x SS fp32 state slice
  __bf16* sq  = (__bf16*)(S + 256 * SS);         // 32 x 256
  __bf16* sk  = sq + CH * DKH;                   // 32 x 256
  __bf16* sw  = sk + CH * DKH;                   // 32 x 256
  float*  su  = (float*)(sw + CH * DKH);         // 32 x SS  (64 cols + 1 pad)
  __bf16* sat = (__bf16*)(su + CH * SS);         // 32 x 32
  __bf16* sun = sat + CH * CH;                   // 64 x SUNK (column-major u_new)
  int bh = blockIdx.y, b = bh >> 2, h = bh & 3;
  int vs0 = blockIdx.x * 64;
  int tid = threadIdx.x, wave = tid >> 5, lane = tid & 31;
  int tm = wave >> 2, tn = wave & 3;             // 2x4 wave grid over 32x64 tiles
  for (int idx = tid; idx < 256 * SS; idx += 256) S[idx] = 0.f;

  for (int ck = 0; ck < NCH; ++ck) {
    size_t gb = ((size_t)bh * L_ + (size_t)ck * CH) * DKH;
    __syncthreads();   // prior chunk fully consumed staging + S updated
#ifdef HAVE_TDM
    if (wave == 0) {   // one wave drives the Tensor Data Mover for all 4 tiles
      tdm_load_2d(lds_off(sq), qn + gb, 1, DKH, DKH, CH, DKH, 0, 0, 0);
      tdm_load_2d(lds_off(sk), kn + gb, 1, DKH, DKH, CH, DKH, 0, 0, 0);
      tdm_load_2d(lds_off(sw), w  + gb, 1, DKH, DKH, CH, DKH, 0, 0, 0);
      // f32 u slice with TDM LDS padding: 1 DWORD pad every 64 DWORDs -> stride SS=65
      tdm_load_2d(lds_off(su), u + gb + vs0, 2, DKH, 64, CH, DKH, 1, 5, 0);
      __builtin_amdgcn_s_wait_tensorcnt(0);
    }
#else
    {
      const uint4* gq = (const uint4*)(qn + gb);
      const uint4* gk = (const uint4*)(kn + gb);
      const uint4* gw = (const uint4*)(w + gb);
      uint4* lq = (uint4*)sq; uint4* lk = (uint4*)sk; uint4* lw = (uint4*)sw;
#pragma unroll
      for (int i = 0; i < 4; ++i) {
        int idx = tid + i * 256;
        lq[idx] = gq[idx]; lk[idx] = gk[idx]; lw[idx] = gw[idx];
      }
      for (int idx = tid; idx < CH * 64; idx += 256) {
        int m = idx >> 6, c = idx & 63;
        su[m * SS + c] = u[gb + (size_t)m * DKH + vs0 + c];
      }
    }
#endif
    __syncthreads();
    // attn_i = (q @ k^T) with causal (incl. diagonal) mask -> bf16
    if (wave < 4) {
      int am = wave >> 1, an = wave & 1;
      v8f acc = zero8();
#pragma unroll
      for (int ks = 0; ks < 8; ++ks) {
        bf16x16 a = ld_a(sq + (am * 16) * DKH + ks * 32, DKH);
        bf16x16 bb = ld_bt(sk, an * 16, ks * 32, DKH);
        acc = wmma_bf16(a, bb, acc);
      }
      int n = an * 16 + (lane & 15), mb = (lane < 16) ? 0 : 8;
#pragma unroll
      for (int e = 0; e < 8; ++e) {
        int m = am * 16 + mb + e;
        sat[m * CH + n] = f2bf((m >= n) ? acc[e] : 0.f);
      }
    }
    __syncthreads();
    // u_new = u - w @ S   (negation folded into B-operand conversion; bf16 WMMA NEG is C-only)
    v8f un = ld_c(su + (tm * 16) * SS + tn * 16, SS);
#pragma unroll
    for (int ks = 0; ks < 8; ++ks) {
      bf16x16 a = ld_a(sw + (tm * 16) * DKH + ks * 32, DKH);
      bf16x16 bb = ld_b_f32(S + (ks * 32) * SS + tn * 16, SS, -1.f);
      un = wmma_bf16(a, bb, un);
    }
    st_cm_bf16(sun + (tn * 16) * SUNK + tm * 16, SUNK, un);  // column-major staging
    __syncthreads();
    // o = q @ S + attn_i @ u_new
    v8f oo = zero8();
#pragma unroll
    for (int ks = 0; ks < 8; ++ks) {
      bf16x16 a = ld_a(sq + (tm * 16) * DKH + ks * 32, DKH);
      bf16x16 bb = ld_b_f32(S + (ks * 32) * SS + tn * 16, SS, 1.f);
      oo = wmma_bf16(a, bb, oo);
    }
    {
      bf16x16 a = ld_a(sat + (tm * 16) * CH, CH);
      bf16x16 bb = ld_bcm(sun + (tn * 16) * SUNK, SUNK);
      oo = wmma_bf16(a, bb, oo);
    }
    {
      int l0 = ck * CH + tm * 16;
      float* op = o + ((size_t)b * L_ + l0) * D_ + h * DKH + vs0 + tn * 16;
      st_c(op, D_, oo);
    }
    __syncthreads();   // all reads of S for this chunk complete before update
    // S += k^T @ u_new  (wave owns dk tiles {2w, 2w+1} x all 4 dv tiles)
#pragma unroll
    for (int t = 0; t < 8; ++t) {
      int td = wave * 2 + (t >> 2), tv = t & 3;
      float* Sp = S + (td * 16) * SS + tv * 16;
      v8f c = ld_c(Sp, SS);
      bf16x16 a = ld_at(sk, td * 16, DKH);
      bf16x16 bb = ld_bcm(sun + (tv * 16) * SUNK, SUNK);
      v8f d = wmma_bf16(a, bb, c);
      st_c(Sp, SS, d);
    }
  }
}

// ---------------- kernel: mix gate + per-head RMS norm -> bf16 ----------------
__global__ __launch_bounds__(256) void mixnorm_k(const float* __restrict__ o,
                                                 const float* __restrict__ vc,
                                                 const float* __restrict__ mixg,
                                                 const float* __restrict__ nw,
                                                 __bf16* __restrict__ on) {
  __shared__ float buf[256];
  int token = blockIdx.x, d = threadIdx.x;
#pragma unroll
  for (int hh = 0; hh < H_; ++hh) {
    size_t gi = (size_t)token * D_ + hh * DKH + d;
    float mx = mixg[token * 4 + hh];
    float ov = mx * o[gi] + (1.f - mx) * vc[gi];
    float ms = block_sum(ov * ov, buf) * (1.f / 256.f);
    on[gi] = f2bf(ov * rsqrtf(ms + 1e-5f) * nw[d]);
  }
}

// ---------------- host-side launch ----------------
extern "C" void kernel_launch(void* const* d_in, const int* in_sizes, int n_in,
                              void* d_out, int out_size, void* d_ws, size_t ws_size,
                              hipStream_t stream) {
  const float* x      = (const float*)d_in[0];
  const float* Wq     = (const float*)d_in[1];
  const float* Wk     = (const float*)d_in[2];
  const float* Wv     = (const float*)d_in[3];
  const float* convq  = (const float*)d_in[4];
  const float* convk  = (const float*)d_in[5];
  const float* convv  = (const float*)d_in[6];
  const float* Wb     = (const float*)d_in[7];
  const float* Wmix   = (const float*)d_in[8];
  const float* norm_w = (const float*)d_in[9];
  const float* Wo     = (const float*)d_in[10];
  float* out = (float*)d_out;

  char* ws = (char*)d_ws;
  size_t off = 0;
  auto alloc = [&](size_t bytes) -> void* {
    void* p = ws + off;
    off = (off + bytes + 255) & ~(size_t)255;
    return p;
  };
  __bf16* xb   = (__bf16*)alloc((size_t)BLT * D_ * 2);
  float*  pre  = (float*)alloc((size_t)BLT * D_ * 4);
  float*  qc   = (float*)alloc((size_t)BLT * D_ * 4);
  float*  kc   = (float*)alloc((size_t)BLT * D_ * 4);
  float*  vc   = (float*)alloc((size_t)BLT * D_ * 4);
  float*  beta = (float*)alloc((size_t)BLT * 4 * 4);
  float*  mixb = (float*)alloc((size_t)BLT * 4 * 4);
  __bf16* qn   = (__bf16*)alloc((size_t)BLT * D_ * 2);
  __bf16* kn   = (__bf16*)alloc((size_t)BLT * D_ * 2);
  __bf16* kb   = (__bf16*)alloc((size_t)BLT * D_ * 2);
  __bf16* vb   = (__bf16*)alloc((size_t)BLT * D_ * 2);
  float*  u    = (float*)alloc((size_t)BLT * D_ * 4);
  __bf16* w    = (__bf16*)alloc((size_t)BLT * D_ * 2);
  float*  ob   = (float*)alloc((size_t)BLT * D_ * 4);
  __bf16* on   = (__bf16*)alloc((size_t)BLT * D_ * 2);
  (void)ws_size; (void)in_sizes; (void)n_in; (void)out_size;

  // 1) x -> bf16 ; gates
  cvt_bf16_k<<<(BLT * D_) / (256 * 4), 256, 0, stream>>>(x, xb, BLT * D_);
  betamix_k<<<BLT, 128, 0, stream>>>(x, Wb, Wmix, beta, mixb);

  // 2) projections (WMMA GEMM) + depthwise conv + silu
  dim3 gg(D_ / BN, BLT / BM);
  dim3 cg(D_ / 256, L_ / CSEG, B_);
  gemm_bf16f32<<<gg, 256, 0, stream>>>(xb, Wq, pre, BLT, D_, D_);
  conv_silu_k<<<cg, 256, 0, stream>>>(pre, convq, qc);
  gemm_bf16f32<<<gg, 256, 0, stream>>>(xb, Wk, pre, BLT, D_, D_);
  conv_silu_k<<<cg, 256, 0, stream>>>(pre, convk, kc);
  gemm_bf16f32<<<gg, 256, 0, stream>>>(xb, Wv, pre, BLT, D_, D_);
  conv_silu_k<<<cg, 256, 0, stream>>>(pre, convv, vc);

  // 3) l2norm + beta scaling, head-major bf16 layout
  norm_split_k<<<BLT, 256, 0, stream>>>(qc, kc, vc, beta, qn, kn, kb, vb);

  // 4) intra-chunk UT transform
  intra_chunk_k<<<dim3(NCH, B_ * H_), 256, 0, stream>>>(kn, kb, vb, u, w);

  // 5) inter-chunk scan; S (256 x 64 fp32) lives in LDS per workgroup
  size_t smem = (size_t)256 * SS * 4 + 3 * (size_t)CH * DKH * 2 +
                (size_t)CH * SS * 4 + (size_t)CH * CH * 2 + (size_t)64 * SUNK * 2;
  scan_chunks_k<<<dim3(DKH / 64, B_ * H_), 256, smem, stream>>>(qn, kn, w, u, ob);

  // 6) mix + RMS norm, final projection into d_out
  mixnorm_k<<<BLT, 256, 0, stream>>>(ob, vc, mixb, norm_w, on);
  gemm_bf16f32<<<gg, 256, 0, stream>>>(on, Wo, out, BLT, D_, D_);
}
